// MultiHeadAttention_30528627540607
// MI455X (gfx1250) — matrix operand order repaired and checked by reading, not compiled, not run on hardware
//
#include <hip/hip_runtime.h>

#define DIMX   1024
#define HEADS_ 16
#define DHEAD  64
#define INNER_ 1024
#define BB     2
#define NN     2048
#define BHN    (BB*HEADS_)

typedef __attribute__((ext_vector_type(16))) __bf16 v16bf;
typedef __attribute__((ext_vector_type(8)))  __bf16 v8bf;
typedef __attribute__((ext_vector_type(8)))  float  v8f;

union V16 { v16bf v; v8bf h[2]; };

// Build a 16-element bf16 A/B operand from two 16-byte chunks.
__device__ __forceinline__ v16bf ld16(const __bf16* p0, const __bf16* p1) {
  V16 u;
  u.h[0] = *(const v8bf*)p0;
  u.h[1] = *(const v8bf*)p1;
  return u.v;
}

__device__ __forceinline__ v8f wmma_bf16(v16bf a, v16bf b, v8f c) {
  return __builtin_amdgcn_wmma_f32_16x16x32_bf16(false, a, false, b, (short)0, c,
                                                 false, false);
}

// ---------------------------------------------------------------------------
// 1) Convert x -> bf16; transpose-convert w_qkv, w_out to [n][k] bf16.
// ---------------------------------------------------------------------------
__global__ void prep_kernel(const float* __restrict__ x,
                            const float* __restrict__ wqkv,
                            const float* __restrict__ wout,
                            __bf16* __restrict__ xb,
                            __bf16* __restrict__ wqkvT,
                            __bf16* __restrict__ woutT) {
  long idx = (long)blockIdx.x * blockDim.x + threadIdx.x;
  const long NX  = (long)BB * NN * DIMX;       // 4M
  const long NW1 = (long)DIMX * 3 * INNER_;    // 3M
  const long NW2 = (long)INNER_ * DIMX;        // 1M
  if (idx < NX) {
    xb[idx] = (__bf16)x[idx];
  } else if (idx < NX + NW1) {
    long t = idx - NX;
    long ncol = t / DIMX;          // 0..3071
    long k    = t % DIMX;
    wqkvT[t] = (__bf16)wqkv[k * (3 * INNER_) + ncol];
  } else if (idx < NX + NW1 + NW2) {
    long t = idx - NX - NW1;
    long ncol = t / INNER_;        // 0..1023
    long k    = t % INNER_;
    woutT[t] = (__bf16)wout[k * DIMX + ncol];
  }
}

// ---------------------------------------------------------------------------
// 2/5) GEMM: C[M x Ncols] f32 = A[M x K] bf16  *  Bt[Ncols x K] bf16 (B pre-T).
// One wave -> 16x64 tile; K stepped by 32; 4 WMMAs per iteration.
// ---------------------------------------------------------------------------
__global__ __launch_bounds__(128) void gemm_bf16(const __bf16* __restrict__ A,
                                                 const __bf16* __restrict__ Bt,
                                                 float* __restrict__ C,
                                                 int M, int Ncols, int K) {
  int gt   = blockIdx.x * blockDim.x + threadIdx.x;
  int wave = gt >> 5;
  int lane = threadIdx.x & 31;
  int ntiles = Ncols >> 6;
  int mt = wave / ntiles;
  int n0 = (wave % ntiles) << 6;
  if ((mt << 4) >= M) return;
  int m  = lane & 15;
  int hi = lane >> 4;

  const __bf16* arow = A + (size_t)((mt << 4) + m) * K;
  const __bf16* b0 = Bt + (size_t)(n0 +  0 + m) * K;
  const __bf16* b1 = Bt + (size_t)(n0 + 16 + m) * K;
  const __bf16* b2 = Bt + (size_t)(n0 + 32 + m) * K;
  const __bf16* b3 = Bt + (size_t)(n0 + 48 + m) * K;

  v8f acc0 = {}, acc1 = {}, acc2 = {}, acc3 = {};
  for (int k0 = 0; k0 < K; k0 += 32) {
    __builtin_prefetch(arow + k0 + 64, 0, 3);
    // A operand: K = 16*hi + e
    v16bf a = ld16(arow + k0 + 16 * hi, arow + k0 + 16 * hi + 8);
    // B operand: K = 16*hi + e (contiguous in pre-transposed Bt)
    v16bf vb0 = ld16(b0 + k0 + 16 * hi, b0 + k0 + 16 * hi + 8);
    v16bf vb1 = ld16(b1 + k0 + 16 * hi, b1 + k0 + 16 * hi + 8);
    v16bf vb2 = ld16(b2 + k0 + 16 * hi, b2 + k0 + 16 * hi + 8);
    v16bf vb3 = ld16(b3 + k0 + 16 * hi, b3 + k0 + 16 * hi + 8);
    acc0 = wmma_bf16(a, vb0, acc0);
    acc1 = wmma_bf16(a, vb1, acc1);
    acc2 = wmma_bf16(a, vb2, acc2);
    acc3 = wmma_bf16(a, vb3, acc3);
  }
  int row0 = (mt << 4) + 8 * hi;  // C layout: VGPR r -> row r + 8*(lane/16)
#pragma unroll
  for (int r = 0; r < 8; ++r) {
    size_t off = (size_t)(row0 + r) * Ncols + n0 + m;
    C[off]      = acc0[r];
    C[off + 16] = acc1[r];
    C[off + 32] = acc2[r];
    C[off + 48] = acc3[r];
  }
}

// ---------------------------------------------------------------------------
// 3) RoPE on q,k + layout change: q,k -> [b,h,n,64] bf16 ; v -> [b,h,64,n] bf16
// ---------------------------------------------------------------------------
__global__ void rope_kernel(const float* __restrict__ qkv,
                            const int* __restrict__ pos,
                            __bf16* __restrict__ qb,
                            __bf16* __restrict__ kb,
                            __bf16* __restrict__ vt) {
  int idx = blockIdx.x * blockDim.x + threadIdx.x;
  if (idx >= BB * NN * INNER_) return;
  int c = idx & (INNER_ - 1);
  int n = (idx >> 10) & (NN - 1);
  int b = idx >> 21;
  int h = c >> 6, d = c & 63;

  const float* row = qkv + (size_t)(b * NN + n) * (3 * INNER_);
  float p  = (float)pos[b * NN + n];
  int   j  = d & 31;
  float inv = __powf(10000.0f, -(float)j * (1.0f / 32.0f));
  float sn, cs;
  __sincosf(p * inv, &sn, &cs);

  int cp = (d < 32) ? (c + 32) : (c - 32);
  float qv = row[c],           qp = row[cp];
  float kv = row[INNER_ + c],  kp = row[INNER_ + cp];
  float qo = (d < 32) ? (qv * cs - qp * sn) : (qv * cs + qp * sn);
  float ko = (d < 32) ? (kv * cs - kp * sn) : (kv * cs + kp * sn);
  float vv = row[2 * INNER_ + c];

  size_t bh = (size_t)b * HEADS_ + h;
  qb[(bh * NN + n) * 64 + d] = (__bf16)qo;
  kb[(bh * NN + n) * 64 + d] = (__bf16)ko;
  vt[(bh * 64 + d) * NN + n] = (__bf16)vv;
}

// ---------------------------------------------------------------------------
// 4) Flash attention. One wave = one (b,h, 16-query tile). Online softmax
// over 32-key blocks. All 4 waves of a block share the same query tile so
// the per-block loop count is uniform (legal __syncthreads).
// ---------------------------------------------------------------------------
__global__ __launch_bounds__(128) void attn_kernel(const __bf16* __restrict__ qb,
                                                   const __bf16* __restrict__ kb,
                                                   const __bf16* __restrict__ vt,
                                                   __bf16* __restrict__ ob) {
  __shared__ __bf16 lds[4][16][32];
  int wid  = threadIdx.x >> 5;
  int lane = threadIdx.x & 31;
  int qt = blockIdx.x >> 3;                 // query tile (shared by block)
  int bh = ((blockIdx.x & 7) << 2) + wid;   // (b*H + h) per wave
  int i0 = qt << 4;
  int m  = lane & 15;
  int hi = lane >> 4;

  // Q as two 16x32 A-operands (d 0..31, 32..63), pre-scaled by 1/sqrt(64).
  const __bf16* qrow = qb + ((size_t)bh * NN + i0 + m) * 64;
  v16bf aq0 = ld16(qrow + 16 * hi,      qrow + 16 * hi + 8);
  v16bf aq1 = ld16(qrow + 32 + 16 * hi, qrow + 32 + 16 * hi + 8);
  const float scale = 0.125f;
#pragma unroll
  for (int e = 0; e < 16; ++e) {
    aq0[e] = (__bf16)((float)aq0[e] * scale);
    aq1[e] = (__bf16)((float)aq1[e] * scale);
  }

  float mrun[8], lrun[8];
#pragma unroll
  for (int r = 0; r < 8; ++r) { mrun[r] = -3.0e38f; lrun[r] = 0.0f; }
  v8f acc0 = {}, acc1 = {}, acc2 = {}, acc3 = {};

  const __bf16* kbase = kb + (size_t)bh * NN * 64;
  const __bf16* vbase = vt + (size_t)bh * 64 * NN;

  for (int j0 = 0; j0 <= i0 + 15; j0 += 32) {
    const __bf16* kL = kbase + (size_t)(j0 + m) * 64;
    const __bf16* kR = kL + 16 * 64;
    __builtin_prefetch(kL + 32 * 64, 0, 3);
    // K B-operands: col = key (lane%16), K-dim = d (16 contiguous per lane half)
    v16bf bL0 = ld16(kL + 16 * hi,      kL + 16 * hi + 8);
    v16bf bL1 = ld16(kL + 32 + 16 * hi, kL + 32 + 16 * hi + 8);
    v16bf bR0 = ld16(kR + 16 * hi,      kR + 16 * hi + 8);
    v16bf bR1 = ld16(kR + 32 + 16 * hi, kR + 32 + 16 * hi + 8);
    v8f s0 = {}, s1 = {};
    s0 = wmma_bf16(aq0, bL0, s0); s0 = wmma_bf16(aq1, bL1, s0);
    s1 = wmma_bf16(aq0, bR0, s1); s1 = wmma_bf16(aq1, bR1, s1);

    // Causal mask + online softmax. Row m=r+8*hi lives across 16 lanes.
    float pl[8], pr[8];
#pragma unroll
    for (int r = 0; r < 8; ++r) {
      int irow = i0 + r + 8 * hi;
      float sl = (j0 + m > irow)      ? -3.0e38f : s0[r];
      float sr = (j0 + 16 + m > irow) ? -3.0e38f : s1[r];
      float mx = fmaxf(sl, sr);
      mx = fmaxf(mx, __shfl_xor(mx, 1, 32));
      mx = fmaxf(mx, __shfl_xor(mx, 2, 32));
      mx = fmaxf(mx, __shfl_xor(mx, 4, 32));
      mx = fmaxf(mx, __shfl_xor(mx, 8, 32));
      float mn   = fmaxf(mrun[r], mx);
      float corr = __expf(mrun[r] - mn);
      float el = __expf(sl - mn);
      float er = __expf(sr - mn);
      float sum = el + er;
      sum += __shfl_xor(sum, 1, 32);
      sum += __shfl_xor(sum, 2, 32);
      sum += __shfl_xor(sum, 4, 32);
      sum += __shfl_xor(sum, 8, 32);
      lrun[r] = lrun[r] * corr + sum;
      mrun[r] = mn;
      pl[r] = el; pr[r] = er;
      acc0[r] *= corr; acc1[r] *= corr; acc2[r] *= corr; acc3[r] *= corr;
    }

    // C-layout -> A-layout reshuffle of P through LDS.
    __syncthreads();
#pragma unroll
    for (int r = 0; r < 8; ++r) {
      lds[wid][r + 8 * hi][m]      = (__bf16)pl[r];
      lds[wid][r + 8 * hi][16 + m] = (__bf16)pr[r];
    }
    __syncthreads();
    v16bf ap = ld16(&lds[wid][m][16 * hi], &lds[wid][m][16 * hi + 8]);

    // P(16x32) @ V(32x64): V pre-transposed so keys are contiguous per lane.
    const __bf16* v0 = vbase + (size_t)(m)      * NN + j0 + 16 * hi;
    const __bf16* v1 = vbase + (size_t)(16 + m) * NN + j0 + 16 * hi;
    const __bf16* v2 = vbase + (size_t)(32 + m) * NN + j0 + 16 * hi;
    const __bf16* v3 = vbase + (size_t)(48 + m) * NN + j0 + 16 * hi;
    v16bf bv0 = ld16(v0, v0 + 8);
    v16bf bv1 = ld16(v1, v1 + 8);
    v16bf bv2 = ld16(v2, v2 + 8);
    v16bf bv3 = ld16(v3, v3 + 8);
    acc0 = wmma_bf16(ap, bv0, acc0);
    acc1 = wmma_bf16(ap, bv1, acc1);
    acc2 = wmma_bf16(ap, bv2, acc2);
    acc3 = wmma_bf16(ap, bv3, acc3);
  }

  int b = bh >> 4;   // HEADS_ == 16
  int h = bh & 15;
#pragma unroll
  for (int r = 0; r < 8; ++r) {
    float invl = 1.0f / lrun[r];
    size_t orow = (size_t)(b * NN + i0 + r + 8 * hi) * INNER_ + h * 64 + m;
    ob[orow]      = (__bf16)(acc0[r] * invl);
    ob[orow + 16] = (__bf16)(acc1[r] * invl);
    ob[orow + 32] = (__bf16)(acc2[r] * invl);
    ob[orow + 48] = (__bf16)(acc3[r] * invl);
  }
}

// ---------------------------------------------------------------------------
extern "C" void kernel_launch(void* const* d_in, const int* in_sizes, int n_in,
                              void* d_out, int out_size, void* d_ws, size_t ws_size,
                              hipStream_t stream) {
  (void)in_sizes; (void)n_in; (void)out_size; (void)ws_size;
  const float* x    = (const float*)d_in[0];
  const int*   pos  = (const int*)d_in[1];
  const float* wqkv = (const float*)d_in[2];
  const float* wout = (const float*)d_in[3];
  float* out = (float*)d_out;

  char* ws = (char*)d_ws;
  const size_t MB = 1024 * 1024;
  __bf16* xb    = (__bf16*)(ws + 0);        //  8 MB: x bf16 [4096][1024]
  __bf16* wqkvT = (__bf16*)(ws + 8  * MB);  //  6 MB: w_qkv^T bf16 [3072][1024]
  __bf16* woutT = (__bf16*)(ws + 14 * MB);  //  2 MB: w_out^T bf16 [1024][1024]
  float*  qkv   = (float*) (ws + 16 * MB);  // 48 MB: qkv f32 [4096][3072]
  __bf16* qb    = (__bf16*)(ws + 64 * MB);  //  8 MB: q bf16 [b,h,n,64]
  __bf16* kbuf  = (__bf16*)(ws + 72 * MB);  //  8 MB: k bf16 [b,h,n,64]
  __bf16* vtb   = (__bf16*)(ws + 80 * MB);  //  8 MB: v bf16 [b,h,64,n]
  __bf16* ob    = (__bf16*)(ws + 88 * MB);  //  8 MB: attn out bf16 [4096][1024]

  // 1) convert + transpose weights
  long totalPrep = (long)BB * NN * DIMX + (long)DIMX * 3 * INNER_ +
                   (long)INNER_ * DIMX;  // 8M elements
  prep_kernel<<<(int)((totalPrep + 255) / 256), 256, 0, stream>>>(
      x, wqkv, wout, xb, wqkvT, woutT);

  // 2) QKV projection: [4096,1024] x [1024,3072]
  {
    int waves = (4096 / 16) * (3072 / 64);  // 12288
    gemm_bf16<<<waves * 32 / 128, 128, 0, stream>>>(xb, wqkvT, qkv,
                                                    4096, 3072, 1024);
  }

  // 3) RoPE + per-head relayout
  rope_kernel<<<(BB * NN * INNER_) / 256, 256, 0, stream>>>(qkv, pos, qb, kbuf, vtb);

  // 4) causal flash attention
  attn_kernel<<<(NN / 16) * 8, 128, 0, stream>>>(qb, kbuf, vtb, ob);

  // 5) output projection: [4096,1024] x [1024,1024] -> d_out f32
  {
    int waves = (4096 / 16) * (1024 / 64);  // 4096
    gemm_bf16<<<waves * 32 / 128, 128, 0, stream>>>(ob, woutT, out,
                                                    4096, 1024, 1024);
  }
}
